// FeedforwardSNNLayer_58437325030132
// MI455X (gfx1250) — matrix operand order repaired and checked by reading, not compiled, not run on hardware
//
#include <hip/hip_runtime.h>
#include <hip/hip_bf16.h>

// ---- CDNA5 vector types ----------------------------------------------------
typedef __attribute__((ext_vector_type(16))) _Float16 v16h;   // WMMA A/B frag
typedef __attribute__((ext_vector_type(8)))  float    v8f;    // WMMA C/D frag
typedef __attribute__((ext_vector_type(4)))  float    f32x4;  // 16B load/store

union AFrag { v16h v; f32x4 f4[2]; };          // 32 bytes = 16 halves
union H8   { f32x4 f4; _Float16 h[8]; };       // 8 halves packed into 16B

__device__ __forceinline__ void cvt_store8(_Float16* dst, f32x4 a, f32x4 b) {
    H8 t;
    t.h[0] = (_Float16)a.x; t.h[1] = (_Float16)a.y;
    t.h[2] = (_Float16)a.z; t.h[3] = (_Float16)a.w;
    t.h[4] = (_Float16)b.x; t.h[5] = (_Float16)b.y;
    t.h[6] = (_Float16)b.z; t.h[7] = (_Float16)b.w;
    *(f32x4*)dst = t.f4;                       // ds_store_b128
}

// ---- Fused SNN layer: mem = own_mems*sigmoid(tau)*(1-own_spikes) + S @ W^T
//      o_spike = heaviside(mem - 0.3); mem = (mem < 0.3) ? mem : 0
// GEMM: C[b,o] = sum_i S[b,i] * W[o,i]  (both K-major => identical tile loads)
// Block tile 128x128, K-stage 64, 8 waves (2 M-waves x 4 N-waves),
// wave tile 64x32 = 4x2 fragments of v_wmma_f32_16x16x32_f16.
// Software-pipelined: next K-stage global loads issue under current WMMAs.
// ----------------------------------------------------------------------------
constexpr int BM = 128, BN = 128, KT = 64, LDK = KT + 8;   // +8 halves pad

__global__ __launch_bounds__(256)
void snn_ff_wmma_kernel(const float* __restrict__ prev,   // [B, K]
                        const float* __restrict__ mems,   // [B, N]
                        const float* __restrict__ spks,   // [B, N]
                        const float* __restrict__ W,      // [N, K]
                        const float* __restrict__ tau,    // [N]
                        float* __restrict__ out,          // [2, B, N]
                        int B, int K, int N)
{
    __shared__ _Float16 Asm[BM][LDK];
    __shared__ _Float16 Bsm[BN][LDK];

    const int tid    = threadIdx.x;
    const int lane   = tid & 31;
    const int wave   = tid >> 5;
    const int wm     = wave >> 2;        // 0..1  -> 64-row slab
    const int wn     = wave & 3;         // 0..3  -> 32-col slab
    const int lane16 = lane & 15;
    const int laneHi = lane >> 4;        // 0/1

    const int rowBase = blockIdx.y * BM;
    const int colBase = blockIdx.x * BN;

    // cooperative tile-load mapping: 2 threads per row, 32 floats each
    const int lrow = tid >> 1;           // 0..127
    const int lcol = (tid & 1) * 32;     // 0 or 32

    v8f acc[4][2];
#pragma unroll
    for (int mi = 0; mi < 4; ++mi)
#pragma unroll
        for (int ni = 0; ni < 2; ++ni)
            acc[mi][ni] = (v8f){0.f,0.f,0.f,0.f,0.f,0.f,0.f,0.f};

    const float* ga = prev + (size_t)(rowBase + lrow) * K + lcol;
    const float* gb = W    + (size_t)(colBase + lrow) * K + lcol;

    // ---- prologue: stage 0 global -> regs ----
    f32x4 av[8], bv[8];
#pragma unroll
    for (int i = 0; i < 8; ++i) {
        av[i] = *(const f32x4*)(ga + i * 4);
        bv[i] = *(const f32x4*)(gb + i * 4);
    }

    for (int kt = 0; kt < K; kt += KT) {
        __syncthreads();                 // previous tile fully consumed
#pragma unroll
        for (int i = 0; i < 4; ++i) {
            cvt_store8(&Asm[lrow][lcol + i * 8], av[2*i], av[2*i+1]);
            cvt_store8(&Bsm[lrow][lcol + i * 8], bv[2*i], bv[2*i+1]);
        }
        __syncthreads();                 // tile visible to all waves

        // speculative prefetch two stages ahead (dropped if untranslatable)
        __builtin_prefetch(ga + kt + 2 * KT, 0, 1);
        __builtin_prefetch(gb + kt + 2 * KT, 0, 1);

        // ---- pipeline: issue next-stage global loads under the WMMAs ----
        if (kt + KT < K) {
#pragma unroll
            for (int i = 0; i < 8; ++i) {
                av[i] = *(const f32x4*)(ga + kt + KT + i * 4);
                bv[i] = *(const f32x4*)(gb + kt + KT + i * 4);
            }
        }

        // ---- compute: 2 wmma K-steps of 32, fragments preloaded ----
#pragma unroll
        for (int ks = 0; ks < KT; ks += 32) {
            AFrag bf[2], af[4];
#pragma unroll
            for (int ni = 0; ni < 2; ++ni) {
                // B frag (32x16): lane holds col N=lane%16, K = 16*(lane/16)..+16
                const _Float16* p = &Bsm[wn*32 + ni*16 + lane16][ks + laneHi*16];
                bf[ni].f4[0] = *(const f32x4*)p;
                bf[ni].f4[1] = *(const f32x4*)(p + 8);
            }
#pragma unroll
            for (int mi = 0; mi < 4; ++mi) {
                // A frag (16x32): lane holds row M=lane%16,
                // K = 16*(lane/16)..+16
                const _Float16* pa = &Asm[wm*64 + mi*16 + lane16][ks + laneHi*16];
                af[mi].f4[0] = *(const f32x4*)pa;
                af[mi].f4[1] = *(const f32x4*)(pa + 8);
            }
#pragma unroll
            for (int mi = 0; mi < 4; ++mi)
#pragma unroll
                for (int ni = 0; ni < 2; ++ni)
                    acc[mi][ni] = __builtin_amdgcn_wmma_f32_16x16x32_f16(
                        false, af[mi].v, false, bf[ni].v,
                        (short)0, acc[mi][ni], false, false);
        }
    }

    // ---- fused epilogue -----------------------------------------------------
    float alpha[2];
#pragma unroll
    for (int ni = 0; ni < 2; ++ni) {
        const int col = colBase + wn*32 + ni*16 + lane16;
        const float t = tau[col];
        alpha[ni] = 1.0f / (1.0f + __expf(-t));
    }

    float* memOut = out;
    float* spkOut = out + (size_t)B * N;

#pragma unroll
    for (int mi = 0; mi < 4; ++mi) {
#pragma unroll
        for (int ni = 0; ni < 2; ++ni) {
            const int col = colBase + wn*32 + ni*16 + lane16;
#pragma unroll
            for (int v = 0; v < 8; ++v) {
                const int row = rowBase + wm*64 + mi*16 + laneHi*8 + v;
                const size_t idx = (size_t)row * N + col;
                const float m = mems[idx] * alpha[ni] * (1.0f - spks[idx])
                              + acc[mi][ni][v];
                spkOut[idx] = (m > 0.3f) ? 1.0f : 0.0f;   // heaviside(m - th)
                memOut[idx] = (m < 0.3f) ? m : 0.0f;      // reset-to-zero
            }
        }
    }
}

extern "C" void kernel_launch(void* const* d_in, const int* in_sizes, int n_in,
                              void* d_out, int out_size, void* d_ws, size_t ws_size,
                              hipStream_t stream) {
    (void)n_in; (void)d_ws; (void)ws_size; (void)out_size;
    const float* prev = (const float*)d_in[0];   // prev_spikes [B, NIN]
    const float* mems = (const float*)d_in[1];   // own_mems    [B, NOUT]
    const float* spks = (const float*)d_in[2];   // own_spikes  [B, NOUT]
    const float* W    = (const float*)d_in[3];   // W           [NOUT, NIN]
    const float* tau  = (const float*)d_in[4];   // tau_m       [NOUT]

    const int NOUT = in_sizes[4];
    const int B    = in_sizes[1] / NOUT;
    const int NIN  = in_sizes[0] / B;

    dim3 grid(NOUT / BN, B / BM);   // (32, 8) for 4096x1024
    snn_ff_wmma_kernel<<<grid, 256, 0, stream>>>(
        prev, mems, spks, W, tau, (float*)d_out, B, NIN, NOUT);
}
